// MultiHeadAttention_39779987096226
// MI455X (gfx1250) — matrix operand order repaired and checked by reading, not compiled, not run on hardware
//
#include <hip/hip_runtime.h>
#include <hip/hip_bf16.h>
#include <cstdint>
#include <cstddef>

#define D_MODEL  768
#define N_HEADS  12
#define HEAD_DIM 64
#define BATCH    4
#define SEQ      2048
#define MROWS    (BATCH * SEQ)   // 8192

typedef __attribute__((ext_vector_type(16))) __bf16 v16bf;
typedef __attribute__((ext_vector_type(8)))  float  v8f;

union FragB16 {
    v16bf          v;
    unsigned short u[16];
    unsigned int   d[8];
};
union FragF32 {
    v8f   v;
    float f[8];
};

static __device__ __forceinline__ unsigned short f32_to_bf16_bits(float x) {
    unsigned int b = __float_as_uint(x);
    unsigned int r = b + 0x7FFFu + ((b >> 16) & 1u);   // round-to-nearest-even
    return (unsigned short)(r >> 16);
}

// pack two f32 into one dword of bf16 (round-half-up)
static __device__ __forceinline__ unsigned int pack_bf16(float lo, float hi) {
    unsigned int a = (__float_as_uint(lo) + 0x8000u) >> 16;
    unsigned int b = (__float_as_uint(hi) + 0x8000u) & 0xFFFF0000u;
    return a | b;
}

// 16 contiguous bf16 (32B) -> fragment, as two b128 loads
static __device__ __forceinline__ void load_run32(FragB16& f, const unsigned short* p) {
    const uint4 t0 = *(const uint4*)(p);
    const uint4 t1 = *(const uint4*)(p + 8);
    f.d[0] = t0.x; f.d[1] = t0.y; f.d[2] = t0.z; f.d[3] = t0.w;
    f.d[4] = t1.x; f.d[5] = t1.y; f.d[6] = t1.z; f.d[7] = t1.w;
}

// A-fragment (16x32 bf16): one contiguous 32B run at k = half*16
static __device__ __forceinline__ void load_afrag(FragB16& f, const unsigned short* row, int half) {
    const uint4 t0 = *(const uint4*)(row + (half << 4));
    const uint4 t1 = *(const uint4*)(row + (half << 4) + 8);
    f.d[0] = t0.x; f.d[1] = t0.y; f.d[2] = t0.z; f.d[3] = t0.w;
    f.d[4] = t1.x; f.d[5] = t1.y; f.d[6] = t1.z; f.d[7] = t1.w;
}

// ---------------------------------------------------------------------------
// K0a: plain f32 -> bf16 conversion (X)
// ---------------------------------------------------------------------------
__global__ void cvt_f32_bf16(const float* __restrict__ in,
                             unsigned short* __restrict__ out, int n) {
    int i = blockIdx.x * blockDim.x + threadIdx.x;
    if (i < n) out[i] = f32_to_bf16_bits(in[i]);
}

// ---------------------------------------------------------------------------
// K0b: f32 [K,N] -> bf16 transposed [N,K]  (weights)
// ---------------------------------------------------------------------------
__global__ void cvt_f32_bf16_T(const float* __restrict__ in,       // [K,N]
                               unsigned short* __restrict__ out) { // [N,K]
    int n = blockIdx.x * blockDim.x + threadIdx.x;
    int k = blockIdx.y;
    if (n < D_MODEL)
        out[(size_t)n * D_MODEL + k] = f32_to_bf16_bits(in[(size_t)k * D_MODEL + n]);
}

// ---------------------------------------------------------------------------
// K1: QKV projection GEMM, M=8192 N=768 K=768, bf16 WMMA, double-buffered.
// mode 0: Q -> [B,H,S,hd] scaled; 1: K -> [B,H,S,hd]; 2: V -> Vt [B,H,hd,S]
// ---------------------------------------------------------------------------
__global__ void __launch_bounds__(128)
qkv_gemm(const unsigned short* __restrict__ Xb,   // [8192,768] bf16
         const unsigned short* __restrict__ Wt,   // [768,768]  bf16 (N x K)
         const float* __restrict__ bias,
         unsigned short* __restrict__ outbuf,
         int mode) {
    const int lane = threadIdx.x & 31;
    const int wave = threadIdx.x >> 5;
    const int half = lane >> 4;
    const int m0   = blockIdx.x * 64 + wave * 16;
    const int n0   = blockIdx.y * 64;

    const unsigned short* arow = Xb + (size_t)(m0 + (lane & 15)) * D_MODEL;
    const unsigned short* wrow[4];
#pragma unroll
    for (int t = 0; t < 4; ++t)
        wrow[t] = Wt + (size_t)(n0 + t * 16 + (lane & 15)) * D_MODEL + (half << 4);

    FragF32 acc[4];
#pragma unroll
    for (int t = 0; t < 4; ++t)
#pragma unroll
        for (int v = 0; v < 8; ++v) acc[t].f[v] = 0.0f;

    FragB16 ab[2], bb[2][4];
    load_afrag(ab[0], arow, half);
#pragma unroll
    for (int t = 0; t < 4; ++t) load_run32(bb[0][t], wrow[t]);

    const int NK = D_MODEL / 32;        // 24 k-steps
#pragma unroll 2
    for (int s = 0; s < NK; ++s) {
        const int cur = s & 1, nxt = cur ^ 1;
        if (s + 1 < NK) {               // prefetch next fragment set
            load_afrag(ab[nxt], arow + (s + 1) * 32, half);
#pragma unroll
            for (int t = 0; t < 4; ++t) load_run32(bb[nxt][t], wrow[t] + (s + 1) * 32);
        }
        // 4 WMMAs share the A block: reuse_a hint must be a literal constant
        acc[0].v = __builtin_amdgcn_wmma_f32_16x16x32_bf16(
            false, ab[cur].v, false, bb[cur][0].v, (short)0, acc[0].v, false, false);
        acc[1].v = __builtin_amdgcn_wmma_f32_16x16x32_bf16(
            false, ab[cur].v, false, bb[cur][1].v, (short)0, acc[1].v, true, false);
        acc[2].v = __builtin_amdgcn_wmma_f32_16x16x32_bf16(
            false, ab[cur].v, false, bb[cur][2].v, (short)0, acc[2].v, true, false);
        acc[3].v = __builtin_amdgcn_wmma_f32_16x16x32_bf16(
            false, ab[cur].v, false, bb[cur][3].v, (short)0, acc[3].v, true, false);
    }

    // Epilogue: bias add + scatter into head-major layouts
#pragma unroll
    for (int t = 0; t < 4; ++t) {
        const int ncol = n0 + t * 16 + (lane & 15);
        const int h  = ncol / HEAD_DIM;
        const int dh = ncol % HEAD_DIM;
        const float bsv = bias[ncol];
#pragma unroll
        for (int v = 0; v < 8; ++v) {
            int m = m0 + (half << 3) + v;       // C/D layout: m = v + 8*(lane/16)
            int b = m / SEQ, s = m % SEQ;
            float val = acc[t].f[v] + bsv;
            size_t idx;
            if (mode == 0) {
                val *= 0.125f;                  // 1/sqrt(HEAD_DIM) folded into Q
                idx = (((size_t)(b * N_HEADS + h)) * SEQ + s) * HEAD_DIM + dh;
            } else if (mode == 1) {
                idx = (((size_t)(b * N_HEADS + h)) * SEQ + s) * HEAD_DIM + dh;
            } else {                            // V transposed
                idx = (((size_t)(b * N_HEADS + h)) * HEAD_DIM + dh) * SEQ + s;
            }
            outbuf[idx] = f32_to_bf16_bits(val);
        }
    }
}

// ---------------------------------------------------------------------------
// K2: fused attention. One workgroup = (b, h, 32 query rows).
// LDS: 128 KB f32 score scratch (16 rows, used twice) + 128 KB bf16 P (32 rows)
//      + 8 KB mask row = 264 KB of the 320 KB WGP budget.
// Two passes of {scores WMMA -> f32 LDS; softmax -> probs(HBM) + bf16 P(LDS)},
// then ctx = P@V with A-fragments read straight from LDS via b128 ds loads.
// ---------------------------------------------------------------------------
__global__ void __launch_bounds__(256)
attn_fused(const unsigned short* __restrict__ Qb,   // [B,H,S,hd] bf16 (scaled)
           const unsigned short* __restrict__ Kb,   // [B,H,S,hd] bf16
           const unsigned short* __restrict__ Vtb,  // [B,H,hd,S] bf16
           const float* __restrict__ mask,          // [B,S]
           float* __restrict__ ctx_out,             // [B,S,D]
           float* __restrict__ probs_out) {         // [B,H,S,S]
    extern __shared__ float smem[];
    float* scf = smem;                                       // [16][SEQ] f32
    unsigned short* Pb = (unsigned short*)(smem + 16 * SEQ); // [32][SEQ] bf16
    float* madd = (float*)(Pb + 32 * SEQ);                   // [SEQ] f32

    const int lane = threadIdx.x & 31;
    const int wave = threadIdx.x >> 5;              // 0..7
    const int half = lane >> 4;
    const int q0   = blockIdx.x * 32;
    const int h    = blockIdx.y;
    const int b    = blockIdx.z;

    const size_t bh = (size_t)(b * N_HEADS + h);
    const unsigned short* Qbase  = Qb  + bh * SEQ * HEAD_DIM;
    const unsigned short* Kbase  = Kb  + bh * SEQ * HEAD_DIM;
    const unsigned short* Vtbase = Vtb + bh * HEAD_DIM * SEQ;

    for (int c = threadIdx.x; c < SEQ; c += 256)
        madd[c] = (1.0f - mask[b * SEQ + c]) * -10000.0f;

    for (int rowoff = 0; rowoff < 32; rowoff += 16) {
        // ---- Phase 1: 16 rows of scores. wave -> 256-col group ----
        {
            FragB16 afr[2];
            const unsigned short* arow =
                Qbase + (size_t)(q0 + rowoff + (lane & 15)) * HEAD_DIM;
            load_afrag(afr[0], arow,      half);
            load_afrag(afr[1], arow + 32, half);

            const int cbase = wave * 256 + (lane & 15);
            FragB16 kb[2][2];
            load_run32(kb[0][0], Kbase + (size_t)cbase * HEAD_DIM + (half << 4));
            load_run32(kb[0][1], Kbase + (size_t)cbase * HEAD_DIM + 32 + (half << 4));
#pragma unroll
            for (int nt = 0; nt < 16; ++nt) {
                const int cur = nt & 1, nxt = cur ^ 1;
                const int ncol = cbase + nt * 16;
                if (nt + 1 < 16) {
                    const unsigned short* krow = Kbase + (size_t)(ncol + 16) * HEAD_DIM;
                    load_run32(kb[nxt][0], krow + (half << 4));
                    load_run32(kb[nxt][1], krow + 32 + (half << 4));
                }
                if (nt < 12)    // stream K ahead into caches (global_prefetch_b8)
                    __builtin_prefetch((const void*)(Kbase + (size_t)(ncol + 64) * HEAD_DIM), 0, 1);
                FragF32 acc;
#pragma unroll
                for (int v = 0; v < 8; ++v) acc.f[v] = 0.0f;
                acc.v = __builtin_amdgcn_wmma_f32_16x16x32_bf16(
                    false, afr[0].v, false, kb[cur][0].v, (short)0, acc.v, false, false);
                acc.v = __builtin_amdgcn_wmma_f32_16x16x32_bf16(
                    false, afr[1].v, false, kb[cur][1].v, (short)0, acc.v, false, false);
#pragma unroll
                for (int v = 0; v < 8; ++v)
                    scf[((half << 3) + v) * SEQ + ncol] = acc.f[v];
            }
        }
        __syncthreads();

        // ---- Phase 2: softmax on 16 rows (2 per wave); pack P once ----
#pragma unroll
        for (int i = 0; i < 2; ++i) {
            const int rl = wave * 2 + i;            // local row 0..15
            const int r  = rowoff + rl;             // row in 0..31
            float4* row4 = (float4*)(scf + rl * SEQ);
            const float4* madd4 = (const float4*)madd;
            float mx = -3.4e38f;
            for (int j = lane; j < SEQ / 4; j += 32) {
                float4 v = row4[j];
                const float4 m = madd4[j];
                v.x += m.x; v.y += m.y; v.z += m.z; v.w += m.w;
                row4[j] = v;
                mx = fmaxf(mx, fmaxf(fmaxf(v.x, v.y), fmaxf(v.z, v.w)));
            }
#pragma unroll
            for (int off = 16; off > 0; off >>= 1)
                mx = fmaxf(mx, __shfl_xor(mx, off, 32));
            float sum = 0.0f;
            for (int j = lane; j < SEQ / 4; j += 32) {
                float4 v = row4[j];
                v.x = __expf(v.x - mx); v.y = __expf(v.y - mx);
                v.z = __expf(v.z - mx); v.w = __expf(v.w - mx);
                row4[j] = v;
                sum += (v.x + v.y) + (v.z + v.w);
            }
#pragma unroll
            for (int off = 16; off > 0; off >>= 1)
                sum += __shfl_xor(sum, off, 32);
            const float inv = 1.0f / sum;
            float4* prow4 = (float4*)(probs_out + (bh * SEQ + (size_t)(q0 + r)) * SEQ);
            uint2*  pb2   = (uint2*)(Pb + (size_t)r * SEQ);
            for (int j = lane; j < SEQ / 4; j += 32) {
                float4 v = row4[j];
                v.x *= inv; v.y *= inv; v.z *= inv; v.w *= inv;
                prow4[j] = v;                                   // probs -> HBM (b128)
                pb2[j] = make_uint2(pack_bf16(v.x, v.y),        // P -> LDS bf16
                                    pack_bf16(v.z, v.w));
            }
        }
        __syncthreads();
    }

    // ---- Phase 3: ctx = P @ V. wave -> (m-tile 0/1, n-tile 0..3) ----
    {
        const int mt2  = wave >> 2;
        const int nt2  = wave & 3;
        const int ncol = nt2 * 16 + (lane & 15);
        const unsigned short* vrow = Vtbase + (size_t)ncol * SEQ;
        const unsigned short* prow = Pb + (size_t)(mt2 * 16 + (lane & 15)) * SEQ;
        FragF32 acc;
#pragma unroll
        for (int v = 0; v < 8; ++v) acc.f[v] = 0.0f;

        FragB16 ab[2], vb[2];
        load_afrag(ab[0], prow, half);                 // ds_load_b128 x2
        load_run32(vb[0], vrow + (half << 4));
        const int NK = SEQ / 32;                       // 64 k-steps
#pragma unroll 4
        for (int ks = 0; ks < NK; ++ks) {
            const int cur = ks & 1, nxt = cur ^ 1;
            if (ks + 1 < NK) {
                load_afrag(ab[nxt], prow + (ks + 1) * 32, half);
                load_run32(vb[nxt], vrow + (ks + 1) * 32 + (half << 4));
            }
            if ((ks & 3) == 0 && ks < 56)
                __builtin_prefetch((const void*)(vrow + (ks + 8) * 32), 0, 1);
            acc.v = __builtin_amdgcn_wmma_f32_16x16x32_bf16(
                false, ab[cur].v, false, vb[cur].v, (short)0, acc.v, false, false);
        }
#pragma unroll
        for (int v = 0; v < 8; ++v) {
            int m = q0 + mt2 * 16 + (half << 3) + v;
            ctx_out[((size_t)b * SEQ + m) * D_MODEL + h * HEAD_DIM + ncol] = acc.f[v];
        }
    }
}

// ---------------------------------------------------------------------------
extern "C" void kernel_launch(void* const* d_in, const int* in_sizes, int n_in,
                              void* d_out, int out_size, void* d_ws, size_t ws_size,
                              hipStream_t stream) {
    const float* X    = (const float*)d_in[0];
    const float* mask = (const float*)d_in[1];
    const float* Wq   = (const float*)d_in[2];
    const float* bq   = (const float*)d_in[3];
    const float* Wk   = (const float*)d_in[4];
    const float* bk   = (const float*)d_in[5];
    const float* Wv   = (const float*)d_in[6];
    const float* bv   = (const float*)d_in[7];

    float* ctx_out   = (float*)d_out;                            // [B,S,D]
    float* probs_out = ctx_out + (size_t)BATCH * SEQ * D_MODEL;  // [B,H,S,S]

    unsigned short* Xb  = (unsigned short*)d_ws;
    unsigned short* Wqt = Xb  + (size_t)MROWS * D_MODEL;
    unsigned short* Wkt = Wqt + (size_t)D_MODEL * D_MODEL;
    unsigned short* Wvt = Wkt + (size_t)D_MODEL * D_MODEL;
    unsigned short* Qb  = Wvt + (size_t)D_MODEL * D_MODEL;
    unsigned short* Kb  = Qb  + (size_t)BATCH * N_HEADS * SEQ * HEAD_DIM;
    unsigned short* Vtb = Kb  + (size_t)BATCH * N_HEADS * SEQ * HEAD_DIM;

    const int nX = MROWS * D_MODEL;
    cvt_f32_bf16<<<(nX + 255) / 256, 256, 0, stream>>>(X, Xb, nX);
    dim3 gT((D_MODEL + 255) / 256, D_MODEL);
    cvt_f32_bf16_T<<<gT, 256, 0, stream>>>(Wq, Wqt);
    cvt_f32_bf16_T<<<gT, 256, 0, stream>>>(Wk, Wkt);
    cvt_f32_bf16_T<<<gT, 256, 0, stream>>>(Wv, Wvt);

    dim3 g1(MROWS / 64, D_MODEL / 64);   // (128, 12)
    qkv_gemm<<<g1, 128, 0, stream>>>(Xb, Wqt, bq, Qb,  0);
    qkv_gemm<<<g1, 128, 0, stream>>>(Xb, Wkt, bk, Kb,  1);
    qkv_gemm<<<g1, 128, 0, stream>>>(Xb, Wvt, bv, Vtb, 2);

    dim3 g2(SEQ / 32, N_HEADS, BATCH);   // (64, 12, 4)
    size_t shmem = (size_t)(16 * SEQ * 4 + 32 * SEQ * 2 + SEQ * 4);  // 264 KB
    attn_fused<<<g2, 256, shmem, stream>>>(Qb, Kb, Vtb, mask, ctx_out, probs_out);
}
